// QbfNewEncoder_28200755265896
// MI455X (gfx1250) — matrix operand order repaired and checked by reading, not compiled, not run on hardware
//
#include <hip/hip_runtime.h>

typedef __bf16 bf16_t;
typedef __attribute__((ext_vector_type(16))) __bf16 v16bf;
typedef __attribute__((ext_vector_type(4)))  __bf16 v4bf;
typedef __attribute__((ext_vector_type(8)))  float  v8f;
typedef __attribute__((ext_vector_type(4)))  float  v4f;
typedef __attribute__((ext_vector_type(4)))  unsigned v4u;
typedef __attribute__((ext_vector_type(8)))  unsigned v8u;

#define NVAR   8000
#define NCLS   15000
#define VROW   416    // row stride of pos/neg var embeddings (32 + 3*128)
#define AVW    288    // row stride of clause aggregation buffer (max D_t)
#define CTW    96     // clause message width (32 clabel + 64 emb)
#define CTROWS 15008  // NCLS padded up to multiple of 32 (K padding, zero-filled)

// ---- WMMA 16x16x32 bf16 fragment layout helpers (contiguous k = 16*(lane/16) + elem; A and B match) ----
// A (16xK=32, 16-bit): lane = m + 16*(k>=16), elem = k&15
__device__ __forceinline__ int a_lane(int m15, int k) { return m15 + ((k >= 16) ? 16 : 0); }
__device__ __forceinline__ int a_elem(int k)          { return k & 15; }
// B (K=32 x 16, 16-bit): lane = n + 16*(k>=16), elem = k&15
__device__ __forceinline__ int b_lane(int n15, int k) { return n15 + ((k >= 16) ? 16 : 0); }

// ============================================================================
// K1: av[c, 0:16*NT] = cmat_pos @ posB + cmat_neg @ negB   (bf16 WMMA, f32 acc)
// Block: 256 threads (8 waves). Tile: M=128 (wave owns one 16-row block),
// N = 16*NT (full width -> cmat streamed from HBM exactly once). K chunk = 32.
// ============================================================================
template <int NT>
__global__ __launch_bounds__(256, 1) void k_clause_gemm(
    const float* __restrict__ cpos, const float* __restrict__ cneg,
    const bf16_t* __restrict__ posB, const bf16_t* __restrict__ negB,
    float* __restrict__ av)
{
  __shared__ alignas(32) bf16_t Af[2][8][32][16];    // [pair][mblock][lane][elem]
  __shared__ alignas(32) bf16_t Bf[2][NT][32][16];   // [pair][ntile][lane][elem]

  const int tid  = threadIdx.x;
  const int lane = tid & 31;
  const int wave = tid >> 5;
  const int cBase = blockIdx.x * 128;

  v8f acc[NT];
  v8f zero8 = {};
#pragma unroll
  for (int i = 0; i < NT; ++i) acc[i] = zero8;

  for (int k0 = 0; k0 < NVAR; k0 += 32) {
    __syncthreads();
    // ---- stage A tiles: 2 pairs x 128 rows x 32 k (fp32 -> bf16), quads on k
#pragma unroll
    for (int i = 0; i < 8; ++i) {
      int e   = tid + 256 * i;        // [0, 2048) quads
      int p   = e >> 10;
      int rem = e & 1023;
      int m   = rem >> 3;             // 0..127
      int k   = (rem & 7) * 4;        // 0,4,...,28
      int row = cBase + m; if (row > NCLS - 1) row = NCLS - 1;
      const float* s = p ? cneg : cpos;
      v4f v = *(const v4f*)&s[(size_t)row * NVAR + k0 + k];
      v4bf b;
      b[0] = (bf16_t)v[0]; b[1] = (bf16_t)v[1]; b[2] = (bf16_t)v[2]; b[3] = (bf16_t)v[3];
      *(v4bf*)&Af[p][m >> 4][a_lane(m & 15, k)][a_elem(k)] = b;
    }
    // ---- stage B tiles: 2 pairs x 32 k x 16*NT n (already bf16), quads on n
#pragma unroll
    for (int i = 0; i < NT; ++i) {
      int e   = tid + 256 * i;        // [0, 256*NT) quads
      int p   = e / (128 * NT);
      int rem = e % (128 * NT);
      int k   = rem / (4 * NT);       // 0..31
      int n   = (rem % (4 * NT)) * 4; // 0..16*NT-4
      const bf16_t* s = p ? negB : posB;
      v4bf v = *(const v4bf*)&s[(size_t)(k0 + k) * VROW + n];
      int nt = n >> 4, n15 = n & 15, kel = k & 15;
      int bl = b_lane(n15, k);
      Bf[p][nt][bl + 0][kel] = v[0];
      Bf[p][nt][bl + 1][kel] = v[1];
      Bf[p][nt][bl + 2][kel] = v[2];
      Bf[p][nt][bl + 3][kel] = v[3];
    }
    // ---- prefetch next A chunk into cache hierarchy
    if (k0 + 32 < NVAR) {
      int m = tid & 127;
      const float* s = (tid & 128) ? cneg : cpos;
      int row = cBase + m; if (row > NCLS - 1) row = NCLS - 1;
      __builtin_prefetch(&s[(size_t)row * NVAR + k0 + 32], 0, 0);
    }
    __syncthreads();
    // ---- compute: wave owns m-block `wave`, loops all N tiles
    v16bf aP = *(const v16bf*)&Af[0][wave][lane][0];
    v16bf aN = *(const v16bf*)&Af[1][wave][lane][0];
#pragma unroll
    for (int nt = 0; nt < NT; ++nt) {
      v16bf bP = *(const v16bf*)&Bf[0][nt][lane][0];
      acc[nt] = __builtin_amdgcn_wmma_f32_16x16x32_bf16(false, aP, false, bP,
                                                        (short)0, acc[nt], false, false);
      v16bf bN = *(const v16bf*)&Bf[1][nt][lane][0];
      acc[nt] = __builtin_amdgcn_wmma_f32_16x16x32_bf16(false, aN, false, bN,
                                                        (short)0, acc[nt], false, false);
    }
  }
  // ---- epilogue: C layout (16x16 f32): row = r + 8*(lane>=16), col = lane&15
  const int mrow0 = cBase + wave * 16 + ((lane >= 16) ? 8 : 0);
  const int ncol0 = lane & 15;
#pragma unroll
  for (int nt = 0; nt < NT; ++nt) {
#pragma unroll
    for (int r = 0; r < 8; ++r) {
      int row = mrow0 + r;
      if (row < NCLS) av[(size_t)row * AVW + nt * 16 + ncol0] = acc[nt][r];
    }
  }
}

// ============================================================================
// K2: c_t = [bf16(clabels), bf16(relu(av @ W_L^T + B_L))]; pad rows -> 0
// ============================================================================
__global__ __launch_bounds__(256) void k_clause_mlp(
    const float* __restrict__ av, const float* __restrict__ clab,
    const float* __restrict__ W, const float* __restrict__ B,
    bf16_t* __restrict__ ct, int Dt)
{
  const int j = threadIdx.x & 63;
  const int c = blockIdx.x * 4 + (threadIdx.x >> 6);  // grid covers CTROWS
  if (c >= NCLS) {                                    // zero K-padding rows
    for (int idx = j; idx < CTW; idx += 64) ct[(size_t)c * CTW + idx] = (bf16_t)0.0f;
    return;
  }
  const float* arow = &av[(size_t)c * AVW];
  const float* wrow = &W[(size_t)j * Dt];
  float h = B[j];
  for (int k = 0; k < Dt; ++k) h += wrow[k] * arow[k];
  h = h > 0.0f ? h : 0.0f;
  ct[(size_t)c * CTW + 32 + j] = (bf16_t)h;
  if (j < 32) ct[(size_t)c * CTW + j] = (bf16_t)clab[(size_t)c * 32 + j];
}

// ============================================================================
// K3: pv = cmat_pos^T @ c_t ; nv = cmat_neg^T @ c_t   (shared B tile, 2 accs)
// A tiles are DMA'd global->LDS by the Tensor Data Mover (TENSOR_LOAD_TO_LDS,
// TENSORcnt), B tile arrives via GLOBAL_LOAD_ASYNC_TO_LDS_B128 (ASYNCcnt);
// both are then repacked into WMMA fragment order (fp32->bf16 for A).
// ============================================================================
__global__ __launch_bounds__(256, 1) void k_var_gemm(
    const float* __restrict__ cpos, const float* __restrict__ cneg,
    const bf16_t* __restrict__ ct,
    float* __restrict__ pv, float* __restrict__ nv)
{
  __shared__ alignas(32) bf16_t Af[2][8][32][16];
  __shared__ alignas(32) bf16_t Bf[6][32][16];
  __shared__ alignas(16) bf16_t Blin[32][CTW];    // linear ct tile (async dest)
  __shared__ alignas(16) float  AlinF[2 * 32 * 128]; // packed TDM tiles (32 KB)

  const int tid  = threadIdx.x;
  const int lane = tid & 31;
  const int wave = tid >> 5;
  const int vBase = blockIdx.x * 128;
  // uniform clamped tile extents (TDM packs the LDS tile with row stride tw)
  const int tw   = (NVAR - vBase) < 128 ? (NVAR - vBase) : 128;

  v8f accP[6], accN[6];
  v8f zero8 = {};
#pragma unroll
  for (int i = 0; i < 6; ++i) { accP[i] = zero8; accN[i] = zero8; }

  for (int k0 = 0; k0 < CTROWS; k0 += 32) {
    const int kmax = (NCLS - k0) < 32 ? (NCLS - k0) : 32;   // uniform, > 0
    __syncthreads();
    // ---- TDM: DMA the two A tiles (kmax x tw fp32, row stride NVAR) to LDS
    if (wave == 0) {
#pragma unroll
      for (int p = 0; p < 2; ++p) {
        const float* gsrc = (p ? cneg : cpos) + (size_t)k0 * NVAR + vBase;
        unsigned long long ga = (unsigned long long)(const void*)gsrc;
        unsigned lds = (unsigned)(size_t)&AlinF[p * 4096];
        v4u g0;
        g0[0] = 1u;                                   // count=1, user mode
        g0[1] = lds;                                  // lds_addr (bytes)
        g0[2] = (unsigned)ga;                         // global_addr[31:0]
        g0[3] = (unsigned)((ga >> 32) & 0x1FFFFFFull) // global_addr[56:32]
              | (2u << 30);                           // type=2 ("image")
        v8u g1;
        g1[0] = 2u << 16;                             // data_size=4B, no mask/flags
        g1[1] = ((unsigned)tw & 0xFFFFu) << 16;       // tensor_dim0[15:0]
        g1[2] = ((unsigned)kmax & 0xFFFFu) << 16;     // dim0[31:16]=0, tensor_dim1[15:0]
        g1[3] = ((unsigned)tw) << 16;                 // dim1[31:16]=0, tile_dim0
        g1[4] = (unsigned)kmax;                       // tile_dim1, tile_dim2=0
        g1[5] = (unsigned)NVAR;                       // tensor_dim0_stride[31:0]
        g1[6] = 0u;                                   // stride0[47:32], stride1[15:0]
        g1[7] = 0u;                                   // stride1[47:16]
        v4u gz = {};                                  // groups 2/3: 2D tensor, unused
        asm volatile("tensor_load_to_lds %0, %1, %2, %3"
                     :: "s"(g0), "s"(g1), "s"(gz), "s"(gz) : "memory");
      }
    }
    // ---- async copy B tile: 32 rows x 96 bf16 = 384 x 16B segments
#pragma unroll
    for (int i = 0; i < 2; ++i) {
      int e = tid + 256 * i;
      if (e < 384) {
        int k = e / 12, seg = e % 12;
        const bf16_t* g = &ct[(size_t)(k0 + k) * CTW + seg * 8];
        unsigned l = (unsigned)(size_t)&Blin[k][seg * 8];  // flat addr low 32b == LDS addr
        asm volatile("global_load_async_to_lds_b128 %0, %1, off"
                     :: "v"(l), "v"(g) : "memory");
      }
    }
    // ---- prefetch next K chunk of A
    {
      int row = k0 + 32 + (tid & 31);
      if (row < NCLS) {
        const float* s = (tid & 128) ? cneg : cpos;
        __builtin_prefetch(&s[(size_t)row * NVAR + vBase + ((tid >> 5) & 3) * 32], 0, 0);
      }
    }
    // ---- drain this wave's async copies / wave0's TDM, then publish LDS
    asm volatile("s_wait_asynccnt 0x0" ::: "memory");
    if (wave == 0) __builtin_amdgcn_s_wait_tensorcnt((short)0);
    __syncthreads();
    // ---- repack A: Alin (packed kmax x tw fp32) -> Af fragments (bf16)
#pragma unroll
    for (int i = 0; i < 8; ++i) {
      int e   = tid + 256 * i;        // [0, 2048) quads
      int p   = e >> 10;
      int rem = e & 1023;
      int k   = rem >> 5;             // 0..31
      int m   = (rem & 31) * 4;       // 0..124
      v4f v;
      if (k < kmax && m < tw) v = *(const v4f*)&AlinF[p * 4096 + k * tw + m];
      else { v4f z = {}; v = z; }
      int kel = a_elem(k);
#pragma unroll
      for (int q = 0; q < 4; ++q) {
        int mm = m + q;
        Af[p][mm >> 4][a_lane(mm & 15, k)][kel] = (bf16_t)v[q];
      }
    }
    // ---- repack B: Blin (row-major) -> Bf (WMMA fragment order)
#pragma unroll
    for (int i = 0; i < 3; ++i) {
      int e = tid + 256 * i;          // [0, 768) quads
      int k = e / 24;                 // 0..31
      int n = (e % 24) * 4;           // 0..92
      v4bf v = *(const v4bf*)&Blin[k][n];
      int nt = n >> 4, kel = k & 15;
      int bl = b_lane(n & 15, k);
      Bf[nt][bl + 0][kel] = v[0];
      Bf[nt][bl + 1][kel] = v[1];
      Bf[nt][bl + 2][kel] = v[2];
      Bf[nt][bl + 3][kel] = v[3];
    }
    __syncthreads();
    v16bf aP = *(const v16bf*)&Af[0][wave][lane][0];
    v16bf aN = *(const v16bf*)&Af[1][wave][lane][0];
#pragma unroll
    for (int nt = 0; nt < 6; ++nt) {
      v16bf b = *(const v16bf*)&Bf[nt][lane][0];
      accP[nt] = __builtin_amdgcn_wmma_f32_16x16x32_bf16(false, aP, false, b,
                                                         (short)0, accP[nt], false, false);
      accN[nt] = __builtin_amdgcn_wmma_f32_16x16x32_bf16(false, aN, false, b,
                                                         (short)0, accN[nt], false, false);
    }
  }
  const int mrow0 = vBase + wave * 16 + ((lane >= 16) ? 8 : 0);
  const int ncol0 = lane & 15;
#pragma unroll
  for (int nt = 0; nt < 6; ++nt) {
#pragma unroll
    for (int r = 0; r < 8; ++r) {
      int row = mrow0 + r;
      if (row < NVAR) {
        size_t o = (size_t)row * CTW + nt * 16 + ncol0;
        pv[o] = accP[nt][r];
        nv[o] = accN[nt][r];
      }
    }
  }
}

// ============================================================================
// K4: relu(pv/nv @ W_C^T + B_C) -> append to pos/neg (swapped) + bf16 mirrors
// ============================================================================
__global__ __launch_bounds__(256) void k_var_mlp(
    const float* __restrict__ pv, const float* __restrict__ nv,
    const float* __restrict__ W, const float* __restrict__ B,
    float* __restrict__ posOut, float* __restrict__ negOut,
    bf16_t* __restrict__ posB, bf16_t* __restrict__ negB, int t)
{
  const int j = threadIdx.x & 63;
  const int v = blockIdx.x * 4 + (threadIdx.x >> 6);  // grid*4 == NVAR
  const float* wrow = &W[(size_t)j * CTW];
  const float* pr = &pv[(size_t)v * CTW];
  const float* nr = &nv[(size_t)v * CTW];
  float hp = B[j], hn = B[j];
  for (int k = 0; k < CTW; ++k) { hp += wrow[k] * pr[k]; hn += wrow[k] * nr[k]; }
  hp = hp > 0.0f ? hp : 0.0f;
  hn = hn > 0.0f ? hn : 0.0f;
  const int colP = 32 + 128 * t + j;
  const int colN = colP + 64;
  const size_t base = (size_t)v * VROW;
  posOut[base + colP] = hp;  posOut[base + colN] = hn;
  negOut[base + colP] = hn;  negOut[base + colN] = hp;
  posB[base + colP] = (bf16_t)hp;  posB[base + colN] = (bf16_t)hn;
  negB[base + colP] = (bf16_t)hn;  negB[base + colN] = (bf16_t)hp;
}

// ============================================================================
// K0: ground labels -> fp32 outputs + bf16 mirrors (cols 0:32)
// ============================================================================
__global__ __launch_bounds__(256) void k_init(
    const float* __restrict__ vl,
    float* __restrict__ posOut, float* __restrict__ negOut,
    bf16_t* __restrict__ posB, bf16_t* __restrict__ negB)
{
  int idx = blockIdx.x * 256 + threadIdx.x;  // NVAR*32 = 256000 exactly
  int v = idx >> 5, d = idx & 31;
  float x = vl[idx];
  size_t o = (size_t)v * VROW + d;
  posOut[o] = x; negOut[o] = x;
  posB[o] = (bf16_t)x; negB[o] = (bf16_t)x;
}

extern "C" void kernel_launch(void* const* d_in, const int* in_sizes, int n_in,
                              void* d_out, int out_size, void* d_ws, size_t ws_size,
                              hipStream_t stream)
{
  (void)in_sizes; (void)out_size; (void)ws_size;
  const float* vlabels = (const float*)d_in[0];
  const float* clabels = (const float*)d_in[1];
  const float* cpos    = (const float*)d_in[2];
  const float* cneg    = (const float*)d_in[3];
  const float *WL[3], *BL[3], *WC[3], *BC[3];
  if (n_in >= 16) {  // tuples expanded to individual inputs
    for (int t = 0; t < 3; ++t) {
      WL[t] = (const float*)d_in[4 + t];
      BL[t] = (const float*)d_in[7 + t];
      WC[t] = (const float*)d_in[10 + t];
      BC[t] = (const float*)d_in[13 + t];
    }
  } else {           // tuples flattened into one buffer each
    const float* wl = (const float*)d_in[4];
    WL[0] = wl; WL[1] = wl + 64 * 32; WL[2] = wl + 64 * 32 + 64 * 160;
    const float* bl = (const float*)d_in[5];
    BL[0] = bl; BL[1] = bl + 64; BL[2] = bl + 128;
    const float* wc = (const float*)d_in[6];
    WC[0] = wc; WC[1] = wc + 64 * 96; WC[2] = wc + 2 * 64 * 96;
    const float* bc = (const float*)d_in[7];
    BC[0] = bc; BC[1] = bc + 64; BC[2] = bc + 128;
  }

  // workspace carve-up (~40 MB)
  char* p = (char*)d_ws;
  auto take = [&](size_t bytes) { char* r = p; p += (bytes + 255) & ~(size_t)255; return r; };
  bf16_t* posB = (bf16_t*)take((size_t)NVAR * VROW * 2);
  bf16_t* negB = (bf16_t*)take((size_t)NVAR * VROW * 2);
  bf16_t* ct   = (bf16_t*)take((size_t)CTROWS * CTW * 2);
  float*  av   = (float*) take((size_t)NCLS * AVW * 4);
  float*  pvb  = (float*) take((size_t)NVAR * CTW * 4);
  float*  nvb  = (float*) take((size_t)NVAR * CTW * 4);

  float* posOut = (float*)d_out;
  float* negOut = posOut + (size_t)NVAR * VROW;

  k_init<<<1000, 256, 0, stream>>>(vlabels, posOut, negOut, posB, negB);
  for (int t = 0; t < 3; ++t) {
    const int Dt = 32 + 128 * t;
    const int g1 = (NCLS + 127) / 128;  // 118
    if (t == 0)      k_clause_gemm<2 ><<<g1, 256, 0, stream>>>(cpos, cneg, posB, negB, av);
    else if (t == 1) k_clause_gemm<10><<<g1, 256, 0, stream>>>(cpos, cneg, posB, negB, av);
    else             k_clause_gemm<18><<<g1, 256, 0, stream>>>(cpos, cneg, posB, negB, av);
    k_clause_mlp<<<CTROWS / 4, 256, 0, stream>>>(av, clabels, WL[t], BL[t], ct, Dt);
    k_var_gemm<<<(NVAR + 127) / 128, 256, 0, stream>>>(cpos, cneg, ct, pvb, nvb);
    k_var_mlp<<<NVAR / 4, 256, 0, stream>>>(pvb, nvb, WC[t], BC[t],
                                            posOut, negOut, posB, negB, t);
  }
}
